// WoodDefectBD_12936441496222
// MI455X (gfx1250) — matrix operand order repaired and checked by reading, not compiled, not run on hardware
//
#include <hip/hip_runtime.h>
#include <hip/hip_bf16.h>

typedef __attribute__((ext_vector_type(16))) __bf16 v16bf;
typedef __attribute__((ext_vector_type(8)))  float  v8f;
typedef __attribute__((ext_vector_type(4)))  unsigned uint32x4;
typedef __attribute__((ext_vector_type(8)))  unsigned uint32x8;

#define BD   4
#define CCH  64
#define HH   256
#define WW   256
#define NN   4096            // 64*64 downsampled tokens
#define BN_EPS 1e-5f
#define SM_SCALE (1.44269504088896340736f * 2.0f)   // log2(e)/T, T=0.5

// ---- Tensor Data Mover: 2D tile Global->LDS (ISA ch.8 D# groups 0/1) ----
// data_size = 2 bytes (bf16). Issued per-wave; tracked by TENSORcnt.
__device__ __forceinline__ void tdm_load_to_lds(unsigned lds_off, const void* gptr,
                                                unsigned tile0, unsigned tile1,
                                                unsigned tdim0, unsigned tdim1,
                                                unsigned stride0) {
    unsigned long long ga = (unsigned long long)(uintptr_t)gptr;
    uint32x4 g0;
    g0.x = 1u;                                   // count=1 (valid user D#)
    g0.y = lds_off;                              // lds_addr (bytes)
    g0.z = (unsigned)ga;                         // global_addr[31:0]
    g0.w = (unsigned)(ga >> 32) | (2u << 30);    // global_addr[56:32] | type=2
    uint32x8 g1;
    g1[0] = 1u << 16;                            // wg_mask=0, data_size=1 (2B)
    g1[1] = (tdim0 & 0xffffu) << 16;             // tensor_dim0[15:0]  @ bits 63:48
    g1[2] = (tdim0 >> 16) | ((tdim1 & 0xffffu) << 16); // tdim0 hi | tdim1 lo
    g1[3] = (tdim1 >> 16) | (tile0 << 16);       // tdim1 hi | tile_dim0
    g1[4] = tile1 & 0xffffu;                     // tile_dim1 ; tile_dim2=0
    g1[5] = stride0;                             // tensor_dim0_stride[31:0]
    g1[6] = 0u;                                  // stride0 hi | dim1_stride lo
    g1[7] = 0u;
    asm volatile("tensor_load_to_lds %0, %1" :: "s"(g0), "s"(g1) : "memory");
}

// ---------------- K0: fold BN into conv weights/bias --------------------
__global__ void wd_prep(const float* __restrict__ w, const float* __restrict__ cb,
                        const float* __restrict__ g,  const float* __restrict__ bt,
                        const float* __restrict__ mn, const float* __restrict__ vr,
                        float* __restrict__ Wf, float* __restrict__ beff) {
    int t = blockIdx.x * 256 + threadIdx.x;          // 64*64*9
    if (t >= 64 * 64 * 9) return;
    int o = t / (64 * 9);
    float scale = g[o] * rsqrtf(vr[o] + BN_EPS);
    Wf[t] = w[t] * scale;
    if ((t % (64 * 9)) == 0) beff[o] = (cb[o] - mn[o]) * scale + bt[o];
}

// -------- K1: 4x bilinear downsample of x -> xs (f32) + V (bf16, C-major)
__global__ void wd_dsx(const float* __restrict__ x, float* __restrict__ xs,
                       __bf16* __restrict__ Vbf) {
    int t = blockIdx.x * 256 + threadIdx.x;          // B*C*N = 2^20
    int n = t & (NN - 1);
    int c = (t >> 12) & 63;
    int b = t >> 18;
    int i = n >> 6, j = n & 63;
    const float* p = x + ((size_t)((b * CCH + c) * HH + 4 * i + 1)) * WW + 4 * j + 1;
    float v = 0.25f * (p[0] + p[1] + p[WW] + p[WW + 1]);
    xs[t] = v;
    Vbf[t] = (__bf16)v;                              // [B][C][N], contiguous write
}

// ---------------- K2: fused conv3x3+BN+ReLU at the 4 sampled positions, avg
__global__ void wd_conv(const float* __restrict__ x, const float* __restrict__ Wf,
                        const float* __restrict__ beff, float* __restrict__ t_ds) {
    __shared__ float lw[4 * 576];
    __shared__ float lb[4];
    int bid = blockIdx.x;                            // b*256 + og*16 + chunk
    int chunk = bid & 15;
    int og = (bid >> 4) & 15;
    int b = bid >> 8;
    int o0 = og * 4;
    for (int e = threadIdx.x; e < 4 * 576; e += 256) lw[e] = Wf[o0 * 576 + e];
    if (threadIdx.x < 4) lb[threadIdx.x] = beff[o0 + threadIdx.x];
    __syncthreads();
    int n = chunk * 256 + threadIdx.x;
    int i = n >> 6, j = n & 63;
    float a[4][4];
#pragma unroll
    for (int o = 0; o < 4; ++o) {
        float bo = lb[o];
        a[o][0] = bo; a[o][1] = bo; a[o][2] = bo; a[o][3] = bo;
    }
    const float* xb = x + ((size_t)(b * CCH) * HH + 4 * i) * WW + 4 * j;
    for (int ic = 0; ic < 64; ++ic) {
        const float* xp = xb + (size_t)ic * HH * WW;
        float4 r0 = *(const float4*)(xp);
        float4 r1 = *(const float4*)(xp + WW);
        float4 r2 = *(const float4*)(xp + 2 * WW);
        float4 r3 = *(const float4*)(xp + 3 * WW);
        float p[4][4] = {{r0.x, r0.y, r0.z, r0.w}, {r1.x, r1.y, r1.z, r1.w},
                         {r2.x, r2.y, r2.z, r2.w}, {r3.x, r3.y, r3.z, r3.w}};
#pragma unroll
        for (int kr = 0; kr < 3; ++kr)
#pragma unroll
            for (int kc = 0; kc < 3; ++kc) {
                float p00 = p[kr][kc],     p01 = p[kr][kc + 1];
                float p10 = p[kr + 1][kc], p11 = p[kr + 1][kc + 1];
#pragma unroll
                for (int o = 0; o < 4; ++o) {
                    float ww = lw[o * 576 + ic * 9 + kr * 3 + kc];  // uniform-addr broadcast
                    a[o][0] = fmaf(ww, p00, a[o][0]);
                    a[o][1] = fmaf(ww, p01, a[o][1]);
                    a[o][2] = fmaf(ww, p10, a[o][2]);
                    a[o][3] = fmaf(ww, p11, a[o][3]);
                }
            }
    }
#pragma unroll
    for (int o = 0; o < 4; ++o) {
        float r = 0.25f * (fmaxf(a[o][0], 0.f) + fmaxf(a[o][1], 0.f) +
                           fmaxf(a[o][2], 0.f) + fmaxf(a[o][3], 0.f));
        t_ds[((size_t)(b * CCH + o0 + o) << 12) + n] = r;
    }
}

// ---------------- K3: L2-normalize over C -> F (bf16, token-major NxC) ----
__global__ void wd_norm(const float* __restrict__ t_ds, __bf16* __restrict__ Fbf) {
    int t = blockIdx.x * 256 + threadIdx.x;          // B*N
    int n = t & (NN - 1);
    int b = t >> 12;
    float v[64];
    float ss = 0.f;
    const float* p = t_ds + (size_t)b * CCH * NN + n;
#pragma unroll
    for (int c = 0; c < 64; ++c) { float q = p[(size_t)c * NN]; v[c] = q; ss += q * q; }
    float inv = 1.0f / fmaxf(sqrtf(ss), 1e-12f);
    __bf16* o = Fbf + (size_t)(b * NN + n) * CCH;
#pragma unroll
    for (int c = 0; c < 64; ++c) o[c] = (__bf16)(v[c] * inv);
}

// ---------------- K4: flash attention (WMMA bf16, f32 accum) + residual ---
// Cosine logits bounded -> no online max; denominator via ones-WMMA.
// K/V tiles staged by the Tensor Data Mover, double-buffered.
// S-tile0 handles EVEN keys, S-tile1 ODD keys so each lane's two P values
// are adjacent in LDS (one packed b32 store); P/V stay natural key order.
__global__ void __launch_bounds__(128) wd_attn(const __bf16* __restrict__ Fbf,
                                               const __bf16* __restrict__ Vbf,
                                               const float* __restrict__ xs,
                                               const float* __restrict__ gamma,
                                               float* __restrict__ out_ds) {
    __shared__ __align__(16) __bf16 Kt[2][32 * 64];   // token-major 32x64
    __shared__ __align__(16) __bf16 Vtr[2][64 * 32];  // channel-major 64x32
    __shared__ __align__(16) __bf16 Pst[4][16 * 32];
    int b     = blockIdx.x >> 6;
    int mblk  = blockIdx.x & 63;
    int wv    = threadIdx.x >> 5;
    int lane  = threadIdx.x & 31;
    int row16 = lane & 15;
    int hi    = lane >> 4;
    int koff  = hi * 8;
    int m0    = mblk * 64 + wv * 16;

    // Q tile 16x64 in A-operand layout (ISA 7.12.2: 16-bit A 16x32, x2 K-chunks)
    const __bf16* qp = Fbf + (size_t)(b * NN + m0 + row16) * CCH;
    v16bf qa0, qa1, onesB;
#pragma unroll
    for (int e = 0; e < 16; ++e) {
        int pp = e >> 1, wb = e & 1;
        int kk = hi * 16 + 2 * pp + wb;
        qa0[e] = qp[kk];
        qa1[e] = qp[32 + kk];
        onesB[e] = (__bf16)1.0f;
    }

    v8f acc[4] = {v8f{}, v8f{}, v8f{}, v8f{}};       // O accumulator: 16 x 64 f32
    v8f accL = {};                                    // softmax denominator rows

    unsigned ktOff0 = (unsigned)(uintptr_t)&Kt[0][0];
    unsigned ktOff1 = (unsigned)(uintptr_t)&Kt[1][0];
    unsigned vtOff0 = (unsigned)(uintptr_t)&Vtr[0][0];
    unsigned vtOff1 = (unsigned)(uintptr_t)&Vtr[1][0];

    auto stage = [&](int nb, int buf) {               // one wave drives the TDM
        if (wv == 0) {
            const void* gk = Fbf + (size_t)(b * NN + nb * 32) * CCH;
            const void* gv = Vbf + (size_t)b * CCH * NN + nb * 32;
            // K: 1-D contiguous 2048 x bf16 (4 KB)
            tdm_load_to_lds(buf ? ktOff1 : ktOff0, gk, 2048u, 1u, 2048u, 1u, 2048u);
            // V: 2-D tile 32 x 64 rows, row stride NN elements (channel-major)
            tdm_load_to_lds(buf ? vtOff1 : vtOff0, gv, 32u, 64u, (unsigned)NN, 64u, (unsigned)NN);
        }
    };

    stage(0, 0);
    for (int nb = 0; nb < NN / 32; ++nb) {
        int buf = nb & 1;
        if (wv == 0) __builtin_amdgcn_s_wait_tensorcnt(0); // DMA for tile nb done
        __syncthreads();                              // publish tile nb
        if (nb + 1 < NN / 32) stage(nb + 1, buf ^ 1); // DMA next under compute
        if (nb + 2 < NN / 32) {                       // warm L2: global_prefetch_b8
            const char* pk = (const char*)(Fbf + (size_t)(b * NN + (nb + 2) * 32) * CCH) + threadIdx.x * 32;
            const char* pv = (const char*)(Vbf + (size_t)b * CCH * NN + (nb + 2) * 32
                                           + (size_t)(threadIdx.x & 63) * NN);
            __builtin_prefetch(pk, 0, 1);
            __builtin_prefetch(pv, 0, 1);
        }
        const __bf16* Kb = Kt[buf];
        const __bf16* Vb = Vtr[buf];

        // S = Q * K^T : tile0 = even keys, tile1 = odd keys; K-dim 64 = 2 WMMAs each
        v8f s0 = {}, s1 = {};
        {
            v16bf kb0, kb1;
#pragma unroll
            for (int e = 0; e < 16; ++e) {
                int pp = e >> 1, wb = e & 1;
                int kk = hi * 16 + 2 * pp + wb;       // B-op: lanes0-15 K0-15, lanes16-31 K16-31
                kb0[e] = Kb[(2 * row16) * 64 + kk];
                kb1[e] = Kb[(2 * row16) * 64 + 32 + kk];
            }
            s0 = __builtin_amdgcn_wmma_f32_16x16x32_bf16(false, qa0, false, kb0, (short)0, s0, false, false);
            s0 = __builtin_amdgcn_wmma_f32_16x16x32_bf16(false, qa1, false, kb1, (short)0, s0, false, false);
#pragma unroll
            for (int e = 0; e < 16; ++e) {
                int pp = e >> 1, wb = e & 1;
                int kk = hi * 16 + 2 * pp + wb;
                kb0[e] = Kb[(2 * row16 + 1) * 64 + kk];
                kb1[e] = Kb[(2 * row16 + 1) * 64 + 32 + kk];
            }
            s1 = __builtin_amdgcn_wmma_f32_16x16x32_bf16(false, qa0, false, kb0, (short)0, s1, false, false);
            s1 = __builtin_amdgcn_wmma_f32_16x16x32_bf16(false, qa1, false, kb1, (short)0, s1, false, false);
        }

        // bounded softmax: P = exp2(S*log2e/T); adjacent even/odd keys pack
        // into one b32 LDS store per row.
#pragma unroll
        for (int r = 0; r < 8; ++r) {
            float p0 = __builtin_amdgcn_exp2f(s0[r] * SM_SCALE);  // key 2*col
            float p1 = __builtin_amdgcn_exp2f(s1[r] * SM_SCALE);  // key 2*col+1
            unsigned short u0 = __builtin_bit_cast(unsigned short, (__bf16)p0);
            unsigned short u1 = __builtin_bit_cast(unsigned short, (__bf16)p1);
            *(unsigned*)&Pst[wv][(r + 8 * hi) * 32 + 2 * row16] =
                (unsigned)u0 | ((unsigned)u1 << 16);
        }

        // reload P in A-operand layout (wave-local LDS round-trip, 2x b128)
        v16bf pa;
#pragma unroll
        for (int e = 0; e < 16; ++e) {
            int pp = e >> 1, wb = e & 1;
            int kk = hi * 16 + 2 * pp + wb;
            pa[e] = Pst[wv][row16 * 32 + kk];
        }

        // denominator rows on the matrix pipe: accL += P * ones
        accL = __builtin_amdgcn_wmma_f32_16x16x32_bf16(false, pa, false, onesB, (short)0, accL, false, false);

        // O += P * V : 4 c-chunks; channel-major V tile -> contiguous b128 reads
#pragma unroll
        for (int cc = 0; cc < 4; ++cc) {
            v16bf vb;
#pragma unroll
            for (int e = 0; e < 16; ++e) {
                int pp = e >> 1, wb = e & 1;
                int kk = hi * 16 + 2 * pp + wb;
                vb[e] = Vb[(cc * 16 + row16) * 32 + kk];
            }
            acc[cc] = __builtin_amdgcn_wmma_f32_16x16x32_bf16(false, pa, false, vb, (short)0, acc[cc], false, false);
        }
    }

    // epilogue: out = xs + gamma * (acc / L)   (every D column of accL = row sum)
    float gm = gamma[0];
#pragma unroll
    for (int cc = 0; cc < 4; ++cc)
#pragma unroll
        for (int r = 0; r < 8; ++r) {
            int row = m0 + r + 8 * hi;
            int c   = cc * 16 + row16;
            float val = acc[cc][r] / accL[r];
            size_t oi = (size_t)(b * CCH + c) * NN + row;
            out_ds[oi] = xs[oi] + gm * val;
        }
}

// ---------------- K5: 4x bilinear upsample back to 256x256 ---------------
__global__ void wd_up(const float* __restrict__ out_ds, float* __restrict__ out) {
    int t  = blockIdx.x * 256 + threadIdx.x;         // B*C*H*W = 2^24
    int xw = t & 255;
    int yh = (t >> 8) & 255;
    int c  = (t >> 16) & 63;
    int b  = t >> 22;
    int ky = yh >> 2, ry = yh & 3;
    int iy0 = ky + ((ry < 2) ? -1 : 0);
    float fy = (ry == 0) ? 0.625f : (ry == 1) ? 0.875f : (ry == 2) ? 0.125f : 0.375f;
    int iy1 = (iy0 + 1 > 63) ? 63 : iy0 + 1;
    iy0 = (iy0 < 0) ? 0 : iy0;
    int kx = xw >> 2, rx = xw & 3;
    int ix0 = kx + ((rx < 2) ? -1 : 0);
    float fx = (rx == 0) ? 0.625f : (rx == 1) ? 0.875f : (rx == 2) ? 0.125f : 0.375f;
    int ix1 = (ix0 + 1 > 63) ? 63 : ix0 + 1;
    ix0 = (ix0 < 0) ? 0 : ix0;
    const float* p = out_ds + (size_t)(b * CCH + c) * NN;
    float v00 = p[iy0 * 64 + ix0], v01 = p[iy0 * 64 + ix1];
    float v10 = p[iy1 * 64 + ix0], v11 = p[iy1 * 64 + ix1];
    out[t] = (1.f - fy) * ((1.f - fx) * v00 + fx * v01) +
             fy         * ((1.f - fx) * v10 + fx * v11);
}

extern "C" void kernel_launch(void* const* d_in, const int* in_sizes, int n_in,
                              void* d_out, int out_size, void* d_ws, size_t ws_size,
                              hipStream_t stream) {
    const float* x      = (const float*)d_in[0];
    const float* conv_w = (const float*)d_in[1];
    const float* conv_b = (const float*)d_in[2];
    const float* bn_g   = (const float*)d_in[3];
    const float* bn_b   = (const float*)d_in[4];
    const float* bn_m   = (const float*)d_in[5];
    const float* bn_v   = (const float*)d_in[6];
    const float* gamma  = (const float*)d_in[7];
    float* out = (float*)d_out;

    char* ws = (char*)d_ws;
    float*  Wf     = (float*)ws;                                   // 147456 B
    float*  beff   = (float*)(ws + 147456);                        // 1024 B pad
    float*  t_ds   = (float*)(ws + 148480);                        // 4 MB
    float*  xs     = (float*)(ws + 148480 + 4194304);              // 4 MB
    float*  out_ds = (float*)(ws + 148480 + 2 * 4194304);          // 4 MB
    __bf16* Fbf    = (__bf16*)(ws + 148480 + 3 * 4194304);         // 2 MB
    __bf16* Vbf    = (__bf16*)(ws + 148480 + 3 * 4194304 + 2097152); // 2 MB

    wd_prep<<<144,  256, 0, stream>>>(conv_w, conv_b, bn_g, bn_b, bn_m, bn_v, Wf, beff);
    wd_dsx <<<4096, 256, 0, stream>>>(x, xs, Vbf);
    wd_conv<<<1024, 256, 0, stream>>>(x, Wf, beff, t_ds);
    wd_norm<<<64,   256, 0, stream>>>(t_ds, Fbf);
    wd_attn<<<256,  128, 0, stream>>>(Fbf, Vbf, xs, gamma, out_ds);
    wd_up  <<<65536, 256, 0, stream>>>(out_ds, out);
}
